// GRU_51067161149599
// MI455X (gfx1250) — matrix operand order repaired and checked by reading, not compiled, not run on hardware
//
#include <hip/hip_runtime.h>
#include <hip/hip_bf16.h>

typedef __attribute__((ext_vector_type(16))) __bf16 v16bf;
typedef __attribute__((ext_vector_type(8)))  float  v8f;

#define B_  32
#define S_  2048
#define E_  256
#define H_  256
#define NT_ 16            // 16-wide N tiles per matrix
#define KS_ 8             // K steps of 32 over K=256
#define FRAG_ELEMS 512    // 32 lanes * 16 bf16 per fragment
#define MAT_PACK (NT_ * KS_ * FRAG_ELEMS)   // 65536 bf16 per packed 256x256 matrix
#define PAD_ 260          // LDS row stride (floats)

static __device__ inline v16bf make_a_frag(float4 f0, float4 f1, float4 f2, float4 f3) {
  v16bf a;
  a[0]=(__bf16)f0.x;  a[1]=(__bf16)f0.y;  a[2]=(__bf16)f0.z;  a[3]=(__bf16)f0.w;
  a[4]=(__bf16)f1.x;  a[5]=(__bf16)f1.y;  a[6]=(__bf16)f1.z;  a[7]=(__bf16)f1.w;
  a[8]=(__bf16)f2.x;  a[9]=(__bf16)f2.y;  a[10]=(__bf16)f2.z; a[11]=(__bf16)f2.w;
  a[12]=(__bf16)f3.x; a[13]=(__bf16)f3.y; a[14]=(__bf16)f3.z; a[15]=(__bf16)f3.w;
  return a;
}

// ---------------------------------------------------------------------------
// Pack W{z,r,h} and U{z,r,h} (f32 [K=256, N=256] row-major) into bf16 WMMA
// B-fragment order: dst[((nt*8 + ks)*32 + lane)*16 + j] = M[ks*32 + 8*(lane>>4) + (j&7) + 16*(j>>3),
//                                                           nt*16 + (lane&15)]
// ---------------------------------------------------------------------------
__global__ void gru_pack_weights(const float* Wz, const float* Wr, const float* Wh,
                                 const float* Uz, const float* Ur, const float* Uh,
                                 __bf16* pw, __bf16* pu) {
  int id = blockIdx.x * blockDim.x + threadIdx.x;      // 2 groups * 3 * MAT_PACK
  int group = id / (3 * MAT_PACK);
  int rem   = id % (3 * MAT_PACK);
  int mat   = rem / MAT_PACK;
  int r2    = rem % MAT_PACK;
  int nt    = r2 / (KS_ * FRAG_ELEMS);
  int r3    = r2 % (KS_ * FRAG_ELEMS);
  int ks    = r3 / FRAG_ELEMS;
  int r4    = r3 % FRAG_ELEMS;
  int lane  = r4 / 16;
  int j     = r4 % 16;
  int n = nt * 16 + (lane & 15);
  int k = ks * 32 + 8 * (lane >> 4) + (j & 7) + ((j >> 3) << 4);
  const float* srcW[3] = {Wz, Wr, Wh};
  const float* srcU[3] = {Uz, Ur, Uh};
  const float* src = group ? srcU[mat] : srcW[mat];
  __bf16* dst = group ? pu : pw;
  dst[rem] = (__bf16)src[k * H_ + n];
}

// ---------------------------------------------------------------------------
// Phase 1: xg[s,b,h] = x[b,s,:] @ Wg  for g in {z,r,h}, stored bf16 [S,B,H].
// Grid: 4096 blocks (b * 128 s-tiles), 256 threads = 8 waves, 6 N-tiles/wave.
// ---------------------------------------------------------------------------
__global__ __launch_bounds__(256) void gru_proj(const float* __restrict__ x,
                                                const __bf16* __restrict__ pw,
                                                __bf16* __restrict__ xz,
                                                __bf16* __restrict__ xr,
                                                __bf16* __restrict__ xh) {
  const int lane = threadIdx.x & 31;
  const int wave = threadIdx.x >> 5;          // 0..7
  const int b    = blockIdx.x >> 7;           // 0..31
  const int s0   = (blockIdx.x & 127) << 4;   // 0..2032
  const int m    = lane & 15;
  const int half = lane >> 4;

  v8f acc[6];
  #pragma unroll
  for (int j = 0; j < 6; j++) acc[j] = (v8f){};

  const float* xrow = x + ((size_t)b * S_ + (s0 + m)) * E_;

  #pragma unroll
  for (int ks = 0; ks < KS_; ks++) {
    int k0 = ks * 32 + 8 * half;
    float4 f0 = *(const float4*)(xrow + k0);
    float4 f1 = *(const float4*)(xrow + k0 + 4);
    float4 f2 = *(const float4*)(xrow + k0 + 16);
    float4 f3 = *(const float4*)(xrow + k0 + 20);
    v16bf a = make_a_frag(f0, f1, f2, f3);
    #pragma unroll
    for (int j = 0; j < 6; j++) {
      int nt = wave * 6 + j;                   // 0..47 across {Wz,Wr,Wh}
      v16bf bf = *(const v16bf*)(pw + ((size_t)(nt * KS_ + ks)) * FRAG_ELEMS + lane * 16);
      acc[j] = __builtin_amdgcn_wmma_f32_16x16x32_bf16(
          false, a, false, bf, (short)0, acc[j], false, false);
    }
  }

  #pragma unroll
  for (int j = 0; j < 6; j++) {
    int nt  = wave * 6 + j;
    int g   = nt >> 4;
    int col = (nt & 15) * 16 + m;
    __bf16* outp = (g == 0) ? xz : (g == 1) ? xr : xh;
    #pragma unroll
    for (int i = 0; i < 8; i++) {
      int s = s0 + i + 8 * half;
      outp[((size_t)s * B_ + b) * H_ + col] = (__bf16)acc[j][i];
    }
  }
}

// ---------------------------------------------------------------------------
// Phase 2: sequential scan. 2 blocks x 512 threads (16 waves). Block handles
// batch rows [16*blockIdx.x, +16). U{z,r,h} B-fragments live in VGPRs for the
// whole 2048-step loop; h[16,256] lives in LDS.
// ---------------------------------------------------------------------------
__global__ __launch_bounds__(512, 1) void gru_scan(const __bf16* __restrict__ xz,
                                                   const __bf16* __restrict__ xr,
                                                   const __bf16* __restrict__ xh,
                                                   const __bf16* __restrict__ pu,
                                                   const float* __restrict__ h0,
                                                   float* __restrict__ out) {
  __shared__ float h_lds[16][PAD_];
  __shared__ float rh_lds[16][PAD_];

  const int tid  = threadIdx.x;
  const int lane = tid & 31;
  const int wave = tid >> 5;                  // 0..15 == N-tile index
  const int b0   = blockIdx.x * 16;
  const int m    = lane & 15;
  const int half = lane >> 4;
  const int col  = wave * 16 + m;

  // Load initial hidden state into LDS.
  for (int i = tid; i < 16 * H_; i += 512) {
    h_lds[i >> 8][i & 255] = h0[(b0 + (i >> 8)) * H_ + (i & 255)];
  }

  // Load persistent B fragments for Uz, Ur, Uh (this wave's 16 output cols).
  v16bf Bz[KS_], Br[KS_], Bh[KS_];
  #pragma unroll
  for (int ks = 0; ks < KS_; ks++) {
    size_t base = ((size_t)(wave * KS_ + ks)) * FRAG_ELEMS + lane * 16;
    Bz[ks] = *(const v16bf*)(pu + 0 * MAT_PACK + base);
    Br[ks] = *(const v16bf*)(pu + 1 * MAT_PACK + base);
    Bh[ks] = *(const v16bf*)(pu + 2 * MAT_PACK + base);
  }
  __syncthreads();

  for (int t = 0; t < S_; t++) {
    const size_t cbase = (size_t)t * (B_ * H_) + (size_t)b0 * H_ + col;

    // C init = input projections (bf16 -> f32).
    v8f az, ar;
    #pragma unroll
    for (int i = 0; i < 8; i++) {
      int row = i + 8 * half;
      az[i] = (float)xz[cbase + (size_t)row * H_];
      ar[i] = (float)xr[cbase + (size_t)row * H_];
    }

    // z/r pre-activations: h @ Uz, h @ Ur (A shared per K-step).
    #pragma unroll
    for (int ks = 0; ks < KS_; ks++) {
      int k0 = ks * 32 + 8 * half;
      float4 f0 = *(const float4*)&h_lds[m][k0];
      float4 f1 = *(const float4*)&h_lds[m][k0 + 4];
      float4 f2 = *(const float4*)&h_lds[m][k0 + 16];
      float4 f3 = *(const float4*)&h_lds[m][k0 + 20];
      v16bf a = make_a_frag(f0, f1, f2, f3);
      az = __builtin_amdgcn_wmma_f32_16x16x32_bf16(false, a, false, Bz[ks], (short)0, az, false, false);
      ar = __builtin_amdgcn_wmma_f32_16x16x32_bf16(false, a, false, Br[ks], (short)0, ar, false, false);
    }

    // Gates + write r*h to LDS for the candidate matmul.
    float holdv[8], zv[8];
    #pragma unroll
    for (int i = 0; i < 8; i++) {
      int row = i + 8 * half;
      float z = 1.0f / (1.0f + __expf(-az[i]));
      float r = 1.0f / (1.0f + __expf(-ar[i]));
      float ho = h_lds[row][col];
      holdv[i] = ho;
      zv[i]    = z;
      rh_lds[row][col] = r * ho;
    }
    __syncthreads();   // rh complete; all h_lds reads of this step done.

    // Candidate pre-activation: xh + (r*h) @ Uh.
    v8f ah;
    #pragma unroll
    for (int i = 0; i < 8; i++) {
      ah[i] = (float)xh[cbase + (size_t)(i + 8 * half) * H_];
    }
    #pragma unroll
    for (int ks = 0; ks < KS_; ks++) {
      int k0 = ks * 32 + 8 * half;
      float4 f0 = *(const float4*)&rh_lds[m][k0];
      float4 f1 = *(const float4*)&rh_lds[m][k0 + 4];
      float4 f2 = *(const float4*)&rh_lds[m][k0 + 16];
      float4 f3 = *(const float4*)&rh_lds[m][k0 + 20];
      v16bf a = make_a_frag(f0, f1, f2, f3);
      ah = __builtin_amdgcn_wmma_f32_16x16x32_bf16(false, a, false, Bh[ks], (short)0, ah, false, false);
    }

    // State update + output store.
    #pragma unroll
    for (int i = 0; i < 8; i++) {
      int row = i + 8 * half;
      float hb = tanhf(ah[i]);
      float hn = (1.0f - zv[i]) * holdv[i] + zv[i] * hb;
      h_lds[row][col] = hn;
      out[((size_t)(b0 + row)) * (S_ * H_) + (size_t)t * H_ + col] = hn;
    }
    __syncthreads();   // h_lds updated before next step's reads.
  }
}

// ---------------------------------------------------------------------------
// Launch: pack -> proj -> scan, all on `stream`.
// Workspace layout (bf16 elems): pw[3*MAT_PACK] | pu[3*MAT_PACK] |
//                                xz[S*B*H] | xr[S*B*H] | xh[S*B*H]  (~97 MB)
// ---------------------------------------------------------------------------
extern "C" void kernel_launch(void* const* d_in, const int* in_sizes, int n_in,
                              void* d_out, int out_size, void* d_ws, size_t ws_size,
                              hipStream_t stream) {
  (void)in_sizes; (void)n_in; (void)out_size; (void)ws_size;
  const float* x  = (const float*)d_in[0];
  const float* h0 = (const float*)d_in[1];
  const float* Wz = (const float*)d_in[2];
  const float* Uz = (const float*)d_in[3];
  const float* Wr = (const float*)d_in[4];
  const float* Ur = (const float*)d_in[5];
  const float* Wh = (const float*)d_in[6];
  const float* Uh = (const float*)d_in[7];
  float* out = (float*)d_out;

  __bf16* pw = (__bf16*)d_ws;
  __bf16* pu = pw + 3 * MAT_PACK;
  __bf16* xz = pu + 3 * MAT_PACK;
  __bf16* xr = xz + (size_t)S_ * B_ * H_;
  __bf16* xh = xr + (size_t)S_ * B_ * H_;

  gru_pack_weights<<<(2 * 3 * MAT_PACK) / 256, 256, 0, stream>>>(Wz, Wr, Wh, Uz, Ur, Uh, pw, pu);
  gru_proj<<<B_ * (S_ / 16), 256, 0, stream>>>(x, pw, xz, xr, xh);
  gru_scan<<<B_ / 16, 512, 0, stream>>>(xz, xr, xh, pu, h0, out);
}
